// LoraQKV_28853590294972
// MI455X (gfx1250) — matrix operand order repaired and checked by reading, not compiled, not run on hardware
//
#include <hip/hip_runtime.h>

// ---------- types ----------
typedef __attribute__((ext_vector_type(16))) __bf16 v16bf;
typedef __attribute__((ext_vector_type(8)))  float  v8f;

union Frag { v16bf v; unsigned u[8]; };

#define SEQ   2048
#define HEADS 32
#define NEGV  -1000000000.0f
#define ASCALE 0.08838834764831845f   // (64+64)^-0.5
#define RMSEPS 1e-6f

__device__ __forceinline__ unsigned short bfbits(float f) {
    unsigned u = __builtin_bit_cast(unsigned, f);
    unsigned r = u + 0x7FFFu + ((u >> 16) & 1u);   // round-to-nearest-even
    return (unsigned short)(r >> 16);
}
__device__ __forceinline__ __bf16 f2bf(float f) {
    return __builtin_bit_cast(__bf16, bfbits(f));
}
__device__ __forceinline__ v8f wmma_bf16(v16bf a, v16bf b, v8f c) {
    return __builtin_amdgcn_wmma_f32_16x16x32_bf16(false, a, false, b, (short)0, c, false, false);
}

// low 32 bits of a generic pointer to LDS == workgroup-relative LDS byte offset
__device__ __forceinline__ unsigned lds_off(const void* p) {
    return (unsigned)(unsigned long long)p;
}
// async DMA: 16B global -> LDS, tracked by ASYNCcnt (bypasses VGPRs)
__device__ __forceinline__ void async_cp16(unsigned lds, const void* gaddr) {
    asm volatile("global_load_async_to_lds_b128 %0, %1, off"
                 :: "v"(lds), "v"(gaddr) : "memory");
}
// same with +16B instruction offset (applied to both LDS and global address)
__device__ __forceinline__ void async_cp16_off16(unsigned lds, const void* gaddr) {
    asm volatile("global_load_async_to_lds_b128 %0, %1, off offset:16"
                 :: "v"(lds), "v"(gaddr) : "memory");
}
__device__ __forceinline__ void async_wait0() {
    asm volatile("s_wait_asynccnt 0" ::: "memory");
}

// ---------- fp32 -> bf16 conversion (n divisible by 1024) ----------
__global__ __launch_bounds__(256) void f32_to_bf16_k(const float* __restrict__ x,
                                                     __bf16* __restrict__ y) {
    int i = (blockIdx.x * 256 + threadIdx.x) * 4;
    float4 v = *(const float4*)(x + i);
    ushort4 o;
    o.x = bfbits(v.x); o.y = bfbits(v.y); o.z = bfbits(v.z); o.w = bfbits(v.w);
    *(ushort4*)((unsigned short*)y + i) = o;
}

// ---------- generic bf16 GEMM: C[M,N] = A[M,K] * B[N,K]^T ----------
// block = 256 threads (8 waves as 4x2), tile 128x64, K-step 32,
// double-buffered LDS filled by async global->LDS DMA
__global__ __launch_bounds__(256) void gemm_bf16_k(const __bf16* __restrict__ A,
                                                   const __bf16* __restrict__ B,
                                                   float* __restrict__ C,
                                                   int M, int N, int K) {
    __shared__ __bf16 As[2][128][32];
    __shared__ __bf16 Bs[2][64][32];
    const int m0 = blockIdx.y * 128, n0 = blockIdx.x * 64;
    const int tid = threadIdx.x;
    const int wave = tid >> 5, lane = tid & 31;
    const int lh = lane >> 4, lr = lane & 15;
    const int wm = wave >> 1, wn = wave & 1;      // 4 x 2 waves, 32x32 each

    v8f acc[2][2] = {};
    const int arow = tid >> 1, ahalf = tid & 1;   // A tile: 128 rows x 2 halves (32B)
    const int brow = tid >> 2, bchk  = tid & 3;   // B tile: 64 rows x 4 chunks (16B)

    auto stage = [&](int buf, int k0) {
        unsigned la = lds_off(&As[buf][arow][ahalf * 16]);
        const void* ga = A + (size_t)(m0 + arow) * K + k0 + ahalf * 16;
        async_cp16(la, ga);
        async_cp16_off16(la, ga);
        unsigned lb = lds_off(&Bs[buf][brow][bchk * 8]);
        const void* gb = B + (size_t)(n0 + brow) * K + k0 + bchk * 8;
        async_cp16(lb, gb);
    };

    stage(0, 0);
    async_wait0();
    __syncthreads();

    int buf = 0;
    for (int k0 = 0; k0 < K; k0 += 32) {
        const int nbuf = buf ^ 1;
        if (k0 + 32 < K) stage(nbuf, k0 + 32);     // prefetch overlaps WMMAs below

        Frag af[2], bfr[2];
#pragma unroll
        for (int mt = 0; mt < 2; ++mt) {
            const unsigned* ar = (const unsigned*)&As[buf][wm * 32 + mt * 16 + lr][0];
#pragma unroll
            for (int d = 0; d < 8; ++d)
                af[mt].u[d] = ar[((d < 4) ? d : d + 4) + 4 * lh];   // A-frag 16x32
        }
#pragma unroll
        for (int nt = 0; nt < 2; ++nt) {
            const unsigned* br = (const unsigned*)&Bs[buf][wn * 32 + nt * 16 + lr][0];
#pragma unroll
            for (int d = 0; d < 8; ++d)
                bfr[nt].u[d] = br[((d < 4) ? d : d + 4) + 4 * lh];   // B-frag 32x16
        }
#pragma unroll
        for (int mt = 0; mt < 2; ++mt)
#pragma unroll
            for (int nt = 0; nt < 2; ++nt)
                acc[mt][nt] = wmma_bf16(af[mt].v, bfr[nt].v, acc[mt][nt]);

        async_wait0();        // prefetched tile fully in LDS
        __syncthreads();      // all waves done reading buf
        buf = nbuf;
    }

#pragma unroll
    for (int mt = 0; mt < 2; ++mt)
#pragma unroll
        for (int nt = 0; nt < 2; ++nt)
#pragma unroll
            for (int v = 0; v < 8; ++v) {
                int row = m0 + wm * 32 + mt * 16 + lh * 8 + v;   // C: M = vgpr + 8*lh
                int col = n0 + wn * 32 + nt * 16 + lr;
                C[(size_t)row * N + col] = acc[mt][nt][v];
            }
}

// ---------- RMS norm (fp32 in, bf16 out), one block per row ----------
__global__ __launch_bounds__(256) void rms_bf16_k(const float* __restrict__ x,
                                                  const float* __restrict__ w,
                                                  __bf16* __restrict__ y,
                                                  int cols, int stride) {
    __shared__ float red[256];
    const int r = blockIdx.x, t = threadIdx.x;
    const float* row = x + (size_t)r * stride;
    float s = 0.f;
    for (int c = t; c < cols; c += 256) { float v = row[c]; s += v * v; }
    red[t] = s;
    __syncthreads();
    for (int o = 128; o > 0; o >>= 1) {
        if (t < o) red[t] += red[t + o];
        __syncthreads();
    }
    float scale = rsqrtf(red[0] / (float)cols + RMSEPS);
    for (int c = t; c < cols; c += 256)
        y[(size_t)r * cols + c] = f2bf(row[c] * scale * w[c]);
}

// ---------- assemble per-head Q/K/V with RoPE ----------
// grid (S, H), 128 threads = feature index
__global__ __launch_bounds__(128) void assemble_k(const float* __restrict__ q,
                                                  const float* __restrict__ kvmat,
                                                  const float* __restrict__ ckv,
                                                  const float* __restrict__ cosb,
                                                  const float* __restrict__ sinb,
                                                  __bf16* __restrict__ Qh,
                                                  __bf16* __restrict__ Kh,
                                                  __bf16* __restrict__ Vh) {
    const int s = blockIdx.x, h = blockIdx.y, d = threadIdx.x;
    const size_t qrow = (size_t)s * 4096 + h * 128;

    float qv;
    if (d < 64) {
        qv = q[qrow + d];
    } else {
        int i = d - 64;
        float x = q[qrow + 64 + i];
        float oth = (i < 32) ? -q[qrow + 64 + i + 32] : q[qrow + 64 + i - 32];
        qv = x * cosb[s * 64 + i] + oth * sinb[s * 64 + i];
    }
    Qh[((size_t)h * SEQ + s) * 128 + d] = f2bf(qv);

    float kvv;
    if (d < 64) {
        kvv = kvmat[qrow + d];                       // k_nope
    } else {
        int i = d - 64;
        float x = ckv[(size_t)s * 960 + 896 + i];    // shared k_rope
        float oth = (i < 32) ? -ckv[(size_t)s * 960 + 896 + i + 32]
                             :  ckv[(size_t)s * 960 + 896 + i - 32];
        kvv = x * cosb[s * 64 + i] + oth * sinb[s * 64 + i];
    }
    Kh[((size_t)h * SEQ + s) * 128 + d] = f2bf(kvv);

    if (d < 64)
        Vh[((size_t)h * SEQ + s) * 64 + d] = f2bf(kvmat[qrow + 64 + d]);
}

// ---------- causal flash attention ----------
// grid (S/128, H), 256 threads = 8 waves x 16 query rows; key blocks of 32,
// double-buffered K (async DMA) + transposed V (manual) staging
__global__ __launch_bounds__(256) void flash_k(const __bf16* __restrict__ Qh,
                                               const __bf16* __restrict__ Kh,
                                               const __bf16* __restrict__ Vh,
                                               float* __restrict__ O) {
    __shared__ __bf16 Ks[2][32][128];     // [buf][key][feat]
    __shared__ __bf16 Vts[2][64][32];     // [buf][d][key] (transposed)
    __shared__ __bf16 Ps[8][16][32];      // per-wave P strip (16 q x 32 k)

    const int qb = blockIdx.x, h = blockIdx.y;
    const int tid = threadIdx.x, wave = tid >> 5, lane = tid & 31;
    const int lh = lane >> 4, lr = lane & 15;
    const int qrow0 = qb * 128 + wave * 16;

    const __bf16* Qbase = Qh + (size_t)h * SEQ * 128;
    const __bf16* Kbase = Kh + (size_t)h * SEQ * 128;
    const __bf16* Vbase = Vh + (size_t)h * SEQ * 64;

    // Q fragments for this wave's 16 rows (16x128 = 4 A-frags), kept in VGPRs
    Frag qf[4];
    {
        const unsigned* qr = (const unsigned*)(Qbase + (size_t)(qrow0 + lr) * 128);
#pragma unroll
        for (int f = 0; f < 4; ++f)
#pragma unroll
            for (int d = 0; d < 8; ++d)
                qf[f].u[d] = qr[f * 16 + ((d < 4) ? d : d + 4) + 4 * lh];
    }

    const int krow = tid >> 3, koff = (tid & 7) * 16;   // K: 32 rows x 2x16B
    const int vkey = tid >> 3, vd0  = (tid & 7) * 8;    // V: 32 keys x 8 d each

    auto stage = [&](int buf, int kb) {
        const int kk0 = kb * 32;
        unsigned lk = lds_off(&Ks[buf][krow][koff]);
        const void* gk = Kbase + (size_t)(kk0 + krow) * 128 + koff;
        async_cp16(lk, gk);
        async_cp16_off16(lk, gk);
        const unsigned short* vs =
            (const unsigned short*)(Vbase + (size_t)(kk0 + vkey) * 64 + vd0);
#pragma unroll
        for (int j = 0; j < 8; ++j)
            Vts[buf][vd0 + j][vkey] = __builtin_bit_cast(__bf16, vs[j]);
    };

    float m8[8], l8[8];
#pragma unroll
    for (int v = 0; v < 8; ++v) { m8[v] = -1e30f; l8[v] = 0.f; }
    v8f o[4] = {};

    const int nkb = qb * 4 + 4;                   // causal: keys up to qb*128+128
    stage(0, 0);
    async_wait0();
    __syncthreads();

    int buf = 0;
    for (int kb = 0; kb < nkb; ++kb) {
        const int kk0 = kb * 32;
        const int nbuf = buf ^ 1;
        if (kb + 1 < nkb) stage(nbuf, kb + 1);    // prefetch next block

        // scores for 16 queries x 32 keys (two 16x16 tiles, K-dim 128 = 4 steps)
        v8f sc[2] = {};
#pragma unroll
        for (int sb = 0; sb < 2; ++sb)
#pragma unroll
            for (int f = 0; f < 4; ++f) {
                Frag kf;
                const unsigned* kr = (const unsigned*)&Ks[buf][sb * 16 + lr][f * 32];
#pragma unroll
                for (int d = 0; d < 8; ++d) kf.u[d] = kr[((d < 4) ? d : d + 4) + 4 * lh];
                sc[sb] = wmma_bf16(qf[f].v, kf.v, sc[sb]);
            }

        // scale + causal mask + row-max over the 32-key strip
        float rmax[8];
#pragma unroll
        for (int v = 0; v < 8; ++v) {
            int row = qrow0 + lh * 8 + v;
            float x0 = sc[0][v] * ASCALE; if (kk0 + lr      > row) x0 = NEGV;
            float x1 = sc[1][v] * ASCALE; if (kk0 + 16 + lr > row) x1 = NEGV;
            sc[0][v] = x0; sc[1][v] = x1;
            float mx = fmaxf(x0, x1);
            mx = fmaxf(mx, __shfl_xor(mx, 1));
            mx = fmaxf(mx, __shfl_xor(mx, 2));
            mx = fmaxf(mx, __shfl_xor(mx, 4));
            mx = fmaxf(mx, __shfl_xor(mx, 8));   // reduce within 16-lane half
            rmax[v] = mx;
        }

        // online softmax update; stash P as bf16 in LDS for A-frag re-layout
#pragma unroll
        for (int v = 0; v < 8; ++v) {
            float mn = fmaxf(m8[v], rmax[v]);
            float corr = __expf(m8[v] - mn);
            m8[v] = mn;
            float p0 = __expf(sc[0][v] - mn);
            float p1 = __expf(sc[1][v] - mn);
            float rs = p0 + p1;
            rs += __shfl_xor(rs, 1);
            rs += __shfl_xor(rs, 2);
            rs += __shfl_xor(rs, 4);
            rs += __shfl_xor(rs, 8);
            l8[v] = l8[v] * corr + rs;
#pragma unroll
            for (int nt = 0; nt < 4; ++nt) o[nt][v] *= corr;
            Ps[wave][lh * 8 + v][lr]      = f2bf(p0);
            Ps[wave][lh * 8 + v][16 + lr] = f2bf(p1);
        }

        // O += P(16x32) x V(32x64)
        Frag pf;
        {
            const unsigned* prow = (const unsigned*)&Ps[wave][lr][0];
#pragma unroll
            for (int d = 0; d < 8; ++d)
                pf.u[d] = prow[((d < 4) ? d : d + 4) + 4 * lh];
        }
#pragma unroll
        for (int nt = 0; nt < 4; ++nt) {
            Frag vf;
            const unsigned* vrow = (const unsigned*)&Vts[buf][nt * 16 + lr][0];
#pragma unroll
            for (int d = 0; d < 8; ++d) vf.u[d] = vrow[((d < 4) ? d : d + 4) + 4 * lh];
            o[nt] = wmma_bf16(pf.v, vf.v, o[nt]);
        }

        async_wait0();
        __syncthreads();
        buf = nbuf;
    }

    // epilogue: normalize and store into [s][h*64+d] fp32
#pragma unroll
    for (int v = 0; v < 8; ++v) {
        int row = qrow0 + lh * 8 + v;
        float inv = 1.0f / l8[v];
        float* orow = O + (size_t)row * 2048 + h * 64;
#pragma unroll
        for (int nt = 0; nt < 4; ++nt)
            orow[nt * 16 + lr] = o[nt][v] * inv;
    }
}

// ---------- host launch ----------
extern "C" void kernel_launch(void* const* d_in, const int* in_sizes, int n_in,
                              void* d_out, int out_size, void* d_ws, size_t ws_size,
                              hipStream_t stream) {
    (void)in_sizes; (void)n_in; (void)out_size; (void)ws_size;

    const float* hidden = (const float*)d_in[0];   // (2048,2048)
    const float* cosb   = (const float*)d_in[1];   // (2048,64)
    const float* sinb   = (const float*)d_in[2];   // (2048,64)
    const float* Wqa    = (const float*)d_in[3];   // (1024,2048)
    const float* q_ln_w = (const float*)d_in[4];   // (1024)
    const float* Wqb    = (const float*)d_in[5];   // (4096,1024)
    const float* Wkva   = (const float*)d_in[6];   // (960,2048)
    const float* kv_ln_w= (const float*)d_in[7];   // (896)
    const float* Wkvb   = (const float*)d_in[8];   // (4096,896)
    const float* Wo     = (const float*)d_in[9];   // (2048,2048)
    float* out = (float*)d_out;                    // (2048,2048)

    char* p = (char*)d_ws;
    auto carve = [&](size_t bytes) -> void* {
        void* r = (void*)p;
        p += (bytes + 255) & ~(size_t)255;
        return r;
    };

    __bf16* hb     = (__bf16*)carve((size_t)2048 * 2048 * 2);
    __bf16* wqab   = (__bf16*)carve((size_t)1024 * 2048 * 2);
    __bf16* wqbb   = (__bf16*)carve((size_t)4096 * 1024 * 2);
    __bf16* wkvab  = (__bf16*)carve((size_t)960  * 2048 * 2);
    __bf16* wkvbb  = (__bf16*)carve((size_t)4096 * 896  * 2);
    __bf16* wob    = (__bf16*)carve((size_t)2048 * 2048 * 2);
    float*  qa_f   = (float*) carve((size_t)2048 * 1024 * 4);
    float*  ckv_f  = (float*) carve((size_t)2048 * 960  * 4);
    __bf16* qan_b  = (__bf16*)carve((size_t)2048 * 1024 * 2);
    __bf16* kvn_b  = (__bf16*)carve((size_t)2048 * 896  * 2);
    float*  q_f    = (float*) carve((size_t)2048 * 4096 * 4);
    float*  kv_f   = (float*) carve((size_t)2048 * 4096 * 4);
    __bf16* Qhb    = (__bf16*)carve((size_t)32 * 2048 * 128 * 2);
    __bf16* Khb    = (__bf16*)carve((size_t)32 * 2048 * 128 * 2);
    __bf16* Vhb    = (__bf16*)carve((size_t)32 * 2048 * 64  * 2);
    float*  attn_f = (float*) carve((size_t)2048 * 2048 * 4);
    __bf16* attn_b = (__bf16*)carve((size_t)2048 * 2048 * 2);

    auto conv = [&](const float* src, __bf16* dst, size_t n) {
        f32_to_bf16_k<<<dim3((unsigned)(n / 1024)), 256, 0, stream>>>(src, dst);
    };

    // 1. precision conversion
    conv(hidden, hb,    (size_t)2048 * 2048);
    conv(Wqa,    wqab,  (size_t)1024 * 2048);
    conv(Wqb,    wqbb,  (size_t)4096 * 1024);
    conv(Wkva,   wkvab, (size_t)960  * 2048);
    conv(Wkvb,   wkvbb, (size_t)4096 * 896);
    conv(Wo,     wob,   (size_t)2048 * 2048);

    // 2. LoRA down-projections
    gemm_bf16_k<<<dim3(1024 / 64, 2048 / 128), 256, 0, stream>>>(hb, wqab,  qa_f,  2048, 1024, 2048);
    gemm_bf16_k<<<dim3(960  / 64, 2048 / 128), 256, 0, stream>>>(hb, wkvab, ckv_f, 2048, 960,  2048);

    // 3. RMS norms (kv norm reads first 896 cols of ckv, stride 960)
    rms_bf16_k<<<2048, 256, 0, stream>>>(qa_f,  q_ln_w,  qan_b, 1024, 1024);
    rms_bf16_k<<<2048, 256, 0, stream>>>(ckv_f, kv_ln_w, kvn_b, 896,  960);

    // 4. up-projections
    gemm_bf16_k<<<dim3(4096 / 64, 2048 / 128), 256, 0, stream>>>(qan_b, wqbb,  q_f,  2048, 4096, 1024);
    gemm_bf16_k<<<dim3(4096 / 64, 2048 / 128), 256, 0, stream>>>(kvn_b, wkvbb, kv_f, 2048, 4096, 896);

    // 5. RoPE + per-head layout
    assemble_k<<<dim3(2048, 32), 128, 0, stream>>>(q_f, kv_f, ckv_f, cosb, sinb, Qhb, Khb, Vhb);

    // 6. causal flash attention
    flash_k<<<dim3(2048 / 128, 32), 256, 0, stream>>>(Qhb, Khb, Vhb, attn_f);

    // 7. output projection
    conv(attn_f, attn_b, (size_t)2048 * 2048);
    gemm_bf16_k<<<dim3(2048 / 64, 2048 / 128), 256, 0, stream>>>(attn_b, wob, out, 2048, 2048, 2048);
}
